// MultiScaleEpisodicModel_35588099014901
// MI455X (gfx1250) — matrix operand order repaired and checked by reading, not compiled, not run on hardware
//
#include <hip/hip_runtime.h>
#include <hip/hip_fp16.h>

typedef __attribute__((ext_vector_type(16))) _Float16 v16h;
typedef __attribute__((ext_vector_type(8)))  float    v8f;
typedef __attribute__((ext_vector_type(2)))  _Float16 h2;

#define B_   256
#define L_   2048
#define NWAVE 4   // waves per block in phase 1 (16 tokens each)

// ---------------------------------------------------------------------------
// Phase 1: embedding gather + FFN + residual + LayerNorm + 3 projections.
// One wave = one 16-token tile; all GEMMs via v_wmma_f32_16x16x32_f16.
// Writes keys[tok][0:64] = concat(sem(32) | fast(16) | slow(16)) in f32.
// ---------------------------------------------------------------------------
__global__ __launch_bounds__(128) void encoder_keys_kernel(
    const int*   __restrict__ seq,   const float* __restrict__ embed,
    const float* __restrict__ w1,    const float* __restrict__ b1,
    const float* __restrict__ w2,    const float* __restrict__ b2,
    const float* __restrict__ ln_g,  const float* __restrict__ ln_b,
    const float* __restrict__ sem_w, const float* __restrict__ sem_b,
    const float* __restrict__ fast_w,const float* __restrict__ fast_b,
    const float* __restrict__ slow_w,const float* __restrict__ slow_b,
    float* __restrict__ keys)
{
  // f16 weights, stored transposed (wt[n][k] = w[k][n]) so B-fragment k-pairs
  // are contiguous 32-bit LDS loads. Row padding spreads banks.
  __shared__ __attribute__((aligned(16))) _Float16 w1t[128][68];   // 17408 B
  __shared__ __attribute__((aligned(16))) _Float16 w2t[64][132];   // 16896 B
  __shared__ __attribute__((aligned(16))) _Float16 pt [64][68];    //  8704 B
  __shared__ float bias1[128];
  __shared__ float bias2[64], lng[64], lnb[64], pb[64];
  __shared__ int   sq[NWAVE][16];
  // per-wave scratch, reused: first as f16 h1[16][132], then as f32 x[16][68]
  __shared__ __attribute__((aligned(16))) float scratch[NWAVE][1088]; // 17408 B

  const int tid  = threadIdx.x;
  const int wave = tid >> 5;
  const int lane = tid & 31;
  const int n16  = lane & 15;
  const int half = lane >> 4;

  // ---- cooperative staging of weights (f32 -> f16, transposed) ----
  for (int i = tid; i < 64 * 128; i += 128) {
    int k = i >> 7, n = i & 127;
    w1t[n][k] = (_Float16)w1[i];
  }
  for (int i = tid; i < 128 * 64; i += 128) {
    int k = i >> 6, n = i & 63;
    w2t[n][k] = (_Float16)w2[i];
  }
  for (int i = tid; i < 64 * 64; i += 128) {
    int k = i >> 6, n = i & 63;
    float v = (n < 32) ? sem_w[k * 32 + n]
            : (n < 48) ? fast_w[k * 16 + (n - 32)]
                       : slow_w[k * 16 + (n - 48)];
    pt[n][k] = (_Float16)v;
  }
  if (tid < 128) bias1[tid] = b1[tid];
  if (tid < 64) {
    bias2[tid] = b2[tid]; lng[tid] = ln_g[tid]; lnb[tid] = ln_b[tid];
    pb[tid] = (tid < 32) ? sem_b[tid]
            : (tid < 48) ? fast_b[tid - 32] : slow_b[tid - 48];
  }
  __syncthreads();

  const int tokbase = blockIdx.x * (NWAVE * 16) + wave * 16;
  if (lane < 16) sq[wave][lane] = seq[tokbase + lane];   // wave-local; DS in-order

  const int tokA = sq[wave][n16];   // token id of this lane's A-fragment row

  // ---- GEMM1: e(16x64) @ w1(64x128), A = embedding gather (f32 -> f16) ----
  v16h a1[2];
  #pragma unroll
  for (int kb = 0; kb < 2; ++kb)
    #pragma unroll
    for (int p = 0; p < 8; ++p) {
      int kk = kb * 32 + ((p & 4) ? 16 : 0) + ((p & 3) << 1) + (half << 3);
      a1[kb][2 * p]     = (_Float16)embed[tokA * 64 + kk];
      a1[kb][2 * p + 1] = (_Float16)embed[tokA * 64 + kk + 1];
    }

  _Float16* h1 = (_Float16*)&scratch[wave][0];   // view: [16][132] f16
  #pragma unroll
  for (int nt = 0; nt < 8; ++nt) {
    v8f d = {};
    #pragma unroll
    for (int kb = 0; kb < 2; ++kb) {
      v16h bf;
      #pragma unroll
      for (int p = 0; p < 8; ++p) {
        h2 pr = *(const h2*)&w1t[nt * 16 + n16][kb * 32 + ((p & 4) ? 16 : 0) + ((p & 3) << 1) + (half << 3)];
        bf[2 * p] = pr.x; bf[2 * p + 1] = pr.y;
      }
      d = __builtin_amdgcn_wmma_f32_16x16x32_f16(false, a1[kb], false, bf,
                                                 (short)0, d, false, false);
    }
    int col = nt * 16 + n16;
    #pragma unroll
    for (int r = 0; r < 8; ++r) {            // relu + f16 store
      int row = r + (half << 3);
      float v = d[r] + bias1[col];
      v = v > 0.f ? v : 0.f;
      h1[row * 132 + col] = (_Float16)v;
    }
  }

  // ---- GEMM2: h1(16x128) @ w2(128x64), + bias + residual ----
  v16h a2[4];                                  // consume h1 fully into regs
  #pragma unroll
  for (int kb = 0; kb < 4; ++kb)
    #pragma unroll
    for (int p = 0; p < 8; ++p) {
      int kk = kb * 32 + ((p & 4) ? 16 : 0) + ((p & 3) << 1) + (half << 3);
      h2 pr = *(const h2*)&h1[n16 * 132 + kk];
      a2[kb][2 * p] = pr.x; a2[kb][2 * p + 1] = pr.y;
    }
  int tokR[8];
  #pragma unroll
  for (int r = 0; r < 8; ++r) tokR[r] = sq[wave][r + (half << 3)];

  float* xln = &scratch[wave][0];              // view: [16][68] f32 (reuses h1)
  #pragma unroll
  for (int nt = 0; nt < 4; ++nt) {
    v8f d = {};
    #pragma unroll
    for (int kb = 0; kb < 4; ++kb) {
      v16h bf;
      #pragma unroll
      for (int p = 0; p < 8; ++p) {
        h2 pr = *(const h2*)&w2t[nt * 16 + n16][kb * 32 + ((p & 4) ? 16 : 0) + ((p & 3) << 1) + (half << 3)];
        bf[2 * p] = pr.x; bf[2 * p + 1] = pr.y;
      }
      d = __builtin_amdgcn_wmma_f32_16x16x32_f16(false, a2[kb], false, bf,
                                                 (short)0, d, false, false);
    }
    int col = nt * 16 + n16;
    #pragma unroll
    for (int r = 0; r < 8; ++r) {
      int row = r + (half << 3);
      float e = embed[tokR[r] * 64 + col];
      xln[row * 68 + col] = d[r] + bias2[col] + e;   // x = e + ff
    }
  }

  // ---- LayerNorm stats: lane owns row n16, cols half*32..+31; pair-combine ----
  float s = 0.f, s2 = 0.f;
  #pragma unroll
  for (int c = 0; c < 32; ++c) {
    float x = xln[n16 * 68 + (half << 5) + c];
    s += x; s2 += x * x;
  }
  s  += __shfl_xor(s, 16, 32);
  s2 += __shfl_xor(s2, 16, 32);
  const float mu = s * (1.f / 64.f);
  const float rs = rsqrtf(s2 * (1.f / 64.f) - mu * mu + 1e-5f);

  // ---- GEMM3: LN fused into A-fragment load; h(16x64) @ proj(64x64) ----
  v16h a3[2];
  #pragma unroll
  for (int kb = 0; kb < 2; ++kb)
    #pragma unroll
    for (int p = 0; p < 8; ++p)
      #pragma unroll
      for (int e = 0; e < 2; ++e) {
        int kk = kb * 32 + ((p & 4) ? 16 : 0) + ((p & 3) << 1) + (half << 3) + e;
        float x  = xln[n16 * 68 + kk];
        float hn = (x - mu) * rs * lng[kk] + lnb[kk];
        a3[kb][2 * p + e] = (_Float16)hn;
      }

  #pragma unroll
  for (int nt = 0; nt < 4; ++nt) {
    v8f d = {};
    #pragma unroll
    for (int kb = 0; kb < 2; ++kb) {
      v16h bf;
      #pragma unroll
      for (int p = 0; p < 8; ++p) {
        h2 pr = *(const h2*)&pt[nt * 16 + n16][kb * 32 + ((p & 4) ? 16 : 0) + ((p & 3) << 1) + (half << 3)];
        bf[2 * p] = pr.x; bf[2 * p + 1] = pr.y;
      }
      d = __builtin_amdgcn_wmma_f32_16x16x32_f16(false, a3[kb], false, bf,
                                                 (short)0, d, false, false);
    }
    int col = nt * 16 + n16;
    #pragma unroll
    for (int r = 0; r < 8; ++r) {
      int row = r + (half << 3);
      keys[(size_t)(tokbase + row) * 64 + col] = d[r] + pb[col];
    }
  }
}

// ---------------------------------------------------------------------------
// Phase 2: sequential delta-rule scan, one workgroup (3 waves) per batch elem.
// lane = matrix row; full step key replicated into per-lane registers from an
// LDS double buffer; wave0 stages step t+1 while everyone computes step t.
// ---------------------------------------------------------------------------
template<int D, int MODE, bool LOADER>
__device__ __forceinline__ void scan_wave(const float* __restrict__ kp,
                                          float (*kbuf)[64], float* call,
                                          const int lane, const int base)
{
  float m[D];
  #pragma unroll
  for (int j = 0; j < D; ++j) m[j] = 0.f;
  const float invL = 1.0f / (float)L_;

  for (int t = 0; t < L_ - 1; ++t) {
    const float* kc = &kbuf[t & 1][base];
    float kreg[D];
    #pragma unroll
    for (int j = 0; j < D; ++j) kreg[j] = kc[j];          // broadcast loads
    const float kv = kc[lane & (D - 1)];                  // k[row] for this lane
    float kk = 0.f, vp = 0.f;
    #pragma unroll
    for (int j = 0; j < D; ++j) {
      kk = fmaf(kreg[j], kreg[j], kk);                    // k . k
      vp = fmaf(m[j], kreg[j], vp);                       // (M k)[row]
    }
    const float dv = kv - vp / (kk + 1e-6f);
    float w;
    if (MODE == 0)      w = 1.0f;
    else if (MODE == 1) w = (float)(t + 1) * invL;
    else                w = sqrtf((float)(t + 1) * invL);
    const float wdv = w * dv;
    #pragma unroll
    for (int j = 0; j < D; ++j) m[j] = fmaf(wdv, kreg[j], m[j]);  // rank-1 update

    if (LOADER) {                                         // wave0 stages t+1
      float*       kn = &kbuf[(t + 1) & 1][0];
      const float* g  = kp + (size_t)(t + 1) * 64;
      kn[lane]      = g[lane];
      kn[lane + 32] = g[lane + 32];
      if (t + 9 < L_)                                     // keep chain in cache
        __builtin_prefetch(&kp[(size_t)(t + 9) * 64 + lane * 2], 0, 0);
    }
    __syncthreads();
  }

  // query with keys[L-1] (already staged into kbuf[(L-1)&1])
  const float* kq = &kbuf[(L_ - 1) & 1][base];
  float c = 0.f;
  #pragma unroll
  for (int j = 0; j < D; ++j) c = fmaf(m[j], kq[j], c);
  if (lane < D) call[base + lane] = c;
}

__global__ __launch_bounds__(96) void scan_readout_kernel(
    const float* __restrict__ keys, const float* __restrict__ out_w,
    const float* __restrict__ out_b, float* __restrict__ out)
{
  __shared__ float kbuf[2][64];
  __shared__ float call[64];
  const int tid  = threadIdx.x;
  const int wave = tid >> 5;
  const int lane = tid & 31;
  const int b    = blockIdx.x;
  const float* kp = keys + (size_t)b * L_ * 64;

  if (tid < 64) kbuf[0][tid] = kp[tid];
  __syncthreads();

  if (wave == 0)      scan_wave<32, 0, true >(kp, kbuf, call, lane, 0);
  else if (wave == 1) scan_wave<16, 1, false>(kp, kbuf, call, lane, 32);
  else                scan_wave<16, 2, false>(kp, kbuf, call, lane, 48);
  __syncthreads();

  if (tid < 64) {                                         // readout GEMV
    float acc = out_b[tid];
    #pragma unroll
    for (int dd = 0; dd < 64; ++dd)
      acc = fmaf(call[dd], out_w[dd * 64 + tid], acc);
    out[(size_t)b * 64 + tid] = acc;
  }
}

// ---------------------------------------------------------------------------
extern "C" void kernel_launch(void* const* d_in, const int* in_sizes, int n_in,
                              void* d_out, int out_size, void* d_ws, size_t ws_size,
                              hipStream_t stream)
{
  (void)in_sizes; (void)n_in; (void)out_size; (void)ws_size;
  const int*   seq    = (const int*)  d_in[0];
  const float* embed  = (const float*)d_in[1];
  const float* w1     = (const float*)d_in[2];
  const float* b1     = (const float*)d_in[3];
  const float* w2     = (const float*)d_in[4];
  const float* b2     = (const float*)d_in[5];
  const float* ln_g   = (const float*)d_in[6];
  const float* ln_b   = (const float*)d_in[7];
  const float* sem_w  = (const float*)d_in[8];
  const float* sem_b  = (const float*)d_in[9];
  const float* fast_w = (const float*)d_in[10];
  const float* fast_b = (const float*)d_in[11];
  const float* slow_w = (const float*)d_in[12];
  const float* slow_b = (const float*)d_in[13];
  const float* out_w  = (const float*)d_in[14];
  const float* out_b  = (const float*)d_in[15];

  float* keysbuf = (float*)d_ws;   // [B][L][64] f32 = 134 MB
  float* out     = (float*)d_out;  // [B][64] f32

  encoder_keys_kernel<<<(B_ * L_) / (NWAVE * 16), NWAVE * 32, 0, stream>>>(
      seq, embed, w1, b1, w2, b2, ln_g, ln_b,
      sem_w, sem_b, fast_w, fast_b, slow_w, slow_b, keysbuf);

  scan_readout_kernel<<<B_, 96, 0, stream>>>(keysbuf, out_w, out_b, out);
}
